// MHAttention_89352499626513
// MI455X (gfx1250) — matrix operand order repaired and checked by reading, not compiled, not run on hardware
//
#include <hip/hip_runtime.h>

#define DIM 1024
#define HEADS 16
#define DEPTH 64
#define BATCH 4
#define SEQ 2048
#define MTOT (BATCH * SEQ) // 8192
#define CHUNK 256          // K-chunk staged in LDS per TDM transfer
#define NCHUNK (DIM / CHUNK)

typedef __attribute__((ext_vector_type(16))) __bf16 v16bf;
typedef __attribute__((ext_vector_type(8)))  float  v8f;
typedef __attribute__((ext_vector_type(4)))  unsigned int v4u;
typedef __attribute__((ext_vector_type(8)))  int v8i;
typedef __attribute__((ext_vector_type(4)))  int v4i;

union Frag {
    v16bf v;
    uint4 u[2];
};

__device__ __forceinline__ unsigned short f32_bf16(float f) {
    unsigned int u = __float_as_uint(f);
    u += 0x7FFFu + ((u >> 16) & 1u); // round to nearest even
    return (unsigned short)(u >> 16);
}

// Compiler barrier: makes the LDS region look externally written (the TDM
// builtin's LDS side effect is not modeled as an IR store, so without this
// the bstage reads get folded to undef).
__device__ __forceinline__ void lds_publish(const void* p) {
    asm volatile("" : : "r"(p) : "memory");
}

// ---------------------------------------------------------------------------
// Tensor Data Mover: load a rows x cols bf16 tile (row stride = strideElems)
// from global into LDS at byte offset lds_off. D# per CDNA5 ISA §8.3/8.4.
// 6-arg builtin form (clang-23 / therock-10.0 headers).
// ---------------------------------------------------------------------------
__device__ __forceinline__ void tdm_load_tile(unsigned lds_off,
                                              const unsigned short* gsrc,
                                              int rows, int cols,
                                              int strideElems) {
    unsigned long long ga = (unsigned long long)(uintptr_t)gsrc;
    v4u g0;
    g0[0] = 1u;                                   // count=1, user mode
    g0[1] = lds_off;                              // lds_addr (bytes)
    g0[2] = (unsigned)ga;                         // global_addr[31:0]
    g0[3] = (unsigned)((ga >> 32) & 0x01FFFFFFu)  // global_addr[56:32]
            | (2u << 30);                         // type=2 (image)
    v8i g1;
    g1[0] = (1 << 16);                            // data_size=1 -> 2 bytes
    g1[1] = (int)(((unsigned)strideElems & 0xFFFFu) << 16); // tensor_dim0[15:0]
    g1[2] = (int)(((unsigned)strideElems >> 16) & 0xFFFFu); // tensor_dim0[31:16], tensor_dim1[15:0]=0
    g1[3] = (int)(1u | ((unsigned)cols << 16));   // tensor_dim1[31:16]=1 (dim1=65536), tile_dim0=cols
    g1[4] = rows;                                 // tile_dim1=rows, tile_dim2=0
    g1[5] = strideElems;                          // tensor_dim0_stride[31:0]
    g1[6] = 0;                                    // stride[47:32]=0, tensor_dim1_stride lo
    g1[7] = 0;
    v4i gz4 = {0, 0, 0, 0};
    v8i gz8 = {0, 0, 0, 0, 0, 0, 0, 0};
    __builtin_amdgcn_tensor_load_to_lds(g0, g1, gz4, gz4, gz8, 0);
}

__global__ void convert_bf16(const float* __restrict__ src,
                             unsigned short* __restrict__ dst, int n) {
    int i = blockIdx.x * blockDim.x + threadIdx.x;
    if (i < n) dst[i] = f32_bf16(src[i]);
}

// ---------------------------------------------------------------------------
// QKV projection: y = x @ W.T + b  -> q [b,h,s,d]*0.125, k [b,h,s,d], v^T [b,h,d,s]
// Wave computes 16x64; block (4 waves) shares a TDM-staged 64xCHUNK W tile.
// ---------------------------------------------------------------------------
__global__ void __launch_bounds__(128)
qkv_gemm(const unsigned short* __restrict__ xb,
         const unsigned short* __restrict__ wqb,
         const unsigned short* __restrict__ wkb,
         const unsigned short* __restrict__ wvb,
         const float* __restrict__ bq,
         const float* __restrict__ bk,
         const float* __restrict__ bv,
         unsigned short* __restrict__ qo,
         unsigned short* __restrict__ ko,
         unsigned short* __restrict__ vto) {
    __shared__ unsigned short bstage[2][64 * CHUNK]; // 2 x 32 KB

    const int lane = threadIdx.x & 31;
    const int wave = threadIdx.x >> 5;
    const int m0 = (blockIdx.x * 4 + wave) * 16;
    const int h  = blockIdx.y;
    const int n0 = h * 64;
    const int mat = blockIdx.z;
    const unsigned short* W = (mat == 0) ? wqb : (mat == 1) ? wkb : wvb;
    const float* bias       = (mat == 0) ? bq  : (mat == 1) ? bk  : bv;

    const int ar = lane & 15;
    const int ak = (lane >> 4) << 3;   // A: halves split K by 8
    const int bn  = lane & 15;
    const int bkk = (lane >> 4) << 3;  // B: halves split K by 8

    const unsigned short* aptr = xb + (size_t)(m0 + ar) * DIM + ak;
    const unsigned short* wtile = W + (size_t)n0 * DIM;

    if (wave == 0) {
        tdm_load_tile(0u, wtile, 64, CHUNK, DIM);
        __builtin_amdgcn_s_wait_tensorcnt(0);
    }
    __syncthreads();
    lds_publish(&bstage[0][0]);

    v8f acc[4] = {};
    for (int c = 0; c < NCHUNK; ++c) {
        if (c + 1 < NCHUNK && wave == 0) {
            tdm_load_tile((unsigned)(((c + 1) & 1) * (64 * CHUNK * 2)),
                          wtile + (c + 1) * CHUNK, 64, CHUNK, DIM);
        }
        const unsigned short* bbuf = &bstage[c & 1][0];
#pragma unroll
        for (int kk = 0; kk < CHUNK; kk += 32) {
            const int k0 = c * CHUNK + kk;
            __builtin_prefetch(aptr + k0 + 256, 0, 0);
            Frag a;
            a.u[0] = *(const uint4*)(aptr + k0);
            a.u[1] = *(const uint4*)(aptr + k0 + 16);
#pragma unroll
            for (int t = 0; t < 4; ++t) {
                const unsigned short* bp =
                    bbuf + (size_t)(t * 16 + bn) * CHUNK + kk + bkk;
                Frag bm;
                bm.u[0] = *(const uint4*)(bp);
                bm.u[1] = *(const uint4*)(bp + 16);
                acc[t] = __builtin_amdgcn_wmma_f32_16x16x32_bf16(
                    false, a.v, false, bm.v, (short)0, acc[t], false, false);
            }
        }
        if (c + 1 < NCHUNK) {
            if (wave == 0) __builtin_amdgcn_s_wait_tensorcnt(0);
            __syncthreads();
            lds_publish(&bstage[0][0]);
        }
    }

    const int cn  = lane & 15;
    const int cmh = (lane >> 4) << 3;
#pragma unroll
    for (int t = 0; t < 4; ++t) {
#pragma unroll
        for (int v = 0; v < 8; ++v) {
            const int m = m0 + v + cmh;
            const int d = t * 16 + cn;
            float val = acc[t][v] + bias[n0 + d];
            const int b = m >> 11;      // / SEQ
            const int s = m & (SEQ - 1);
            if (mat == 0) {
                val *= 0.125f; // 1/sqrt(64) folded into Q
                qo[(((size_t)(b * HEADS + h)) * SEQ + s) * DEPTH + d] = f32_bf16(val);
            } else if (mat == 1) {
                ko[(((size_t)(b * HEADS + h)) * SEQ + s) * DEPTH + d] = f32_bf16(val);
            } else {
                vto[(((size_t)(b * HEADS + h)) * DEPTH + d) * SEQ + s] = f32_bf16(val);
            }
        }
    }
}

// ---------------------------------------------------------------------------
// Causal flash attention. Wave = 16 query rows; key tiles of 32; online softmax.
// ---------------------------------------------------------------------------
__global__ void __launch_bounds__(128)
flash_attn(const unsigned short* __restrict__ qb,
           const unsigned short* __restrict__ kb,
           const unsigned short* __restrict__ vtb,
           unsigned short* __restrict__ attb) {
    __shared__ unsigned short plds[4][16 * 32];

    const int lane = threadIdx.x & 31;
    const int wave = threadIdx.x >> 5;
    const int m0 = blockIdx.x * 64 + wave * 16; // query row base
    const int bh = blockIdx.y;
    const int b = bh >> 4, h = bh & 15;

    const unsigned short* Qb = qb  + ((size_t)bh * SEQ) * DEPTH;
    const unsigned short* Kb = kb  + ((size_t)bh * SEQ) * DEPTH;
    const unsigned short* Vt = vtb + ((size_t)bh * DEPTH) * SEQ;

    const int ar = lane & 15;
    const int ak = (lane >> 4) << 3;
    const int bn  = lane & 15;
    const int bkk = (lane >> 4) << 3;

    // Q A-fragments for d=0..31 and d=32..63, kept in registers all loop
    Frag q0, q1;
    {
        const unsigned short* qp = Qb + (size_t)(m0 + ar) * DEPTH + ak;
        q0.u[0] = *(const uint4*)(qp);
        q0.u[1] = *(const uint4*)(qp + 16);
        q1.u[0] = *(const uint4*)(qp + 32);
        q1.u[1] = *(const uint4*)(qp + 48);
    }

    v8f O[4] = {};
    float mrow[8], lrow[8];
#pragma unroll
    for (int v = 0; v < 8; ++v) { mrow[v] = -1e30f; lrow[v] = 0.0f; }

    const int cn   = lane & 15;
    const int rmh  = (lane >> 4) << 3;
    const int nkt  = ((m0 + 15) >> 5) + 1; // causal key-tile count

    for (int kt = 0; kt < nkt; ++kt) {
        const int k0 = kt * 32;
        // scores: Q(16x64) x K^T(64x32) -> two 16x16 C frags
        v8f sc[2] = {};
#pragma unroll
        for (int nt = 0; nt < 2; ++nt) {
            const unsigned short* kp =
                Kb + (size_t)(k0 + nt * 16 + bn) * DEPTH + bkk;
            Frag kb0, kb1;
            kb0.u[0] = *(const uint4*)(kp);
            kb0.u[1] = *(const uint4*)(kp + 16);
            kb1.u[0] = *(const uint4*)(kp + 32);
            kb1.u[1] = *(const uint4*)(kp + 48);
            sc[nt] = __builtin_amdgcn_wmma_f32_16x16x32_bf16(
                false, q0.v, false, kb0.v, (short)0, sc[nt], false, false);
            sc[nt] = __builtin_amdgcn_wmma_f32_16x16x32_bf16(
                false, q1.v, false, kb1.v, (short)0, sc[nt], false, false);
        }

        // causal mask + online softmax (rows live across 16-lane halves)
#pragma unroll
        for (int v = 0; v < 8; ++v) {
            const int row = m0 + v + rmh;
            float s0 = (k0 + cn      <= row) ? sc[0][v] : -1e30f;
            float s1 = (k0 + 16 + cn <= row) ? sc[1][v] : -1e30f;
            float tmax = fmaxf(s0, s1);
#pragma unroll
            for (int off = 1; off < 16; off <<= 1)
                tmax = fmaxf(tmax, __shfl_xor(tmax, off, 32));
            const float mnew = fmaxf(mrow[v], tmax);
            const float p0 = __expf(s0 - mnew);
            const float p1 = __expf(s1 - mnew);
            const float alpha = __expf(mrow[v] - mnew);
            float rs = p0 + p1;
#pragma unroll
            for (int off = 1; off < 16; off <<= 1)
                rs += __shfl_xor(rs, off, 32);
            lrow[v] = lrow[v] * alpha + rs;
            mrow[v] = mnew;
#pragma unroll
            for (int t = 0; t < 4; ++t) O[t][v] *= alpha;
            // stage P (C-layout -> LDS row-major 16x32)
            const int prow = v + rmh;
            plds[wave][prow * 32 + cn]      = f32_bf16(p0);
            plds[wave][prow * 32 + 16 + cn] = f32_bf16(p1);
        }

        // reload P in A-fragment layout from LDS
        Frag p;
        {
            const unsigned short* pp = &plds[wave][ar * 32 + ak];
            p.u[0] = *(const uint4*)(pp);
            p.u[1] = *(const uint4*)(pp + 16);
        }

        // O(16x64) += P(16x32) x V(32x64); V stored transposed -> contiguous B
#pragma unroll
        for (int t = 0; t < 4; ++t) {
            const unsigned short* vp =
                Vt + (size_t)(t * 16 + bn) * SEQ + k0 + bkk;
            Frag vb;
            vb.u[0] = *(const uint4*)(vp);
            vb.u[1] = *(const uint4*)(vp + 16);
            O[t] = __builtin_amdgcn_wmma_f32_16x16x32_bf16(
                false, p.v, false, vb.v, (short)0, O[t], false, false);
        }
    }

    // normalize and store att in [b, s, h, d] (== [b,s,DIM]) as bf16
#pragma unroll
    for (int v = 0; v < 8; ++v) {
        const float inv = 1.0f / lrow[v];
        const int m = m0 + v + rmh;
#pragma unroll
        for (int t = 0; t < 4; ++t) {
            const int d = t * 16 + cn;
            attb[(((size_t)b * SEQ + m) * HEADS + h) * DEPTH + d] =
                f32_bf16(O[t][v] * inv);
        }
    }
}

// ---------------------------------------------------------------------------
// Output projection: out = att @ Wo.T + bo (f32 out), TDM-staged Wo tile.
// ---------------------------------------------------------------------------
__global__ void __launch_bounds__(128)
out_gemm(const unsigned short* __restrict__ ab,
         const unsigned short* __restrict__ wob,
         const float* __restrict__ bo,
         float* __restrict__ out) {
    __shared__ unsigned short bstage[2][64 * CHUNK];

    const int lane = threadIdx.x & 31;
    const int wave = threadIdx.x >> 5;
    const int m0 = (blockIdx.x * 4 + wave) * 16;
    const int n0 = blockIdx.y * 64;

    const int ar = lane & 15;
    const int ak = (lane >> 4) << 3;
    const int bn  = lane & 15;
    const int bkk = (lane >> 4) << 3;

    const unsigned short* aptr = ab + (size_t)(m0 + ar) * DIM + ak;
    const unsigned short* wtile = wob + (size_t)n0 * DIM;

    if (wave == 0) {
        tdm_load_tile(0u, wtile, 64, CHUNK, DIM);
        __builtin_amdgcn_s_wait_tensorcnt(0);
    }
    __syncthreads();
    lds_publish(&bstage[0][0]);

    v8f acc[4] = {};
    for (int c = 0; c < NCHUNK; ++c) {
        if (c + 1 < NCHUNK && wave == 0) {
            tdm_load_tile((unsigned)(((c + 1) & 1) * (64 * CHUNK * 2)),
                          wtile + (c + 1) * CHUNK, 64, CHUNK, DIM);
        }
        const unsigned short* bbuf = &bstage[c & 1][0];
#pragma unroll
        for (int kk = 0; kk < CHUNK; kk += 32) {
            const int k0 = c * CHUNK + kk;
            __builtin_prefetch(aptr + k0 + 256, 0, 0);
            Frag a;
            a.u[0] = *(const uint4*)(aptr + k0);
            a.u[1] = *(const uint4*)(aptr + k0 + 16);
#pragma unroll
            for (int t = 0; t < 4; ++t) {
                const unsigned short* bp =
                    bbuf + (size_t)(t * 16 + bn) * CHUNK + kk + bkk;
                Frag bm;
                bm.u[0] = *(const uint4*)(bp);
                bm.u[1] = *(const uint4*)(bp + 16);
                acc[t] = __builtin_amdgcn_wmma_f32_16x16x32_bf16(
                    false, a.v, false, bm.v, (short)0, acc[t], false, false);
            }
        }
        if (c + 1 < NCHUNK) {
            if (wave == 0) __builtin_amdgcn_s_wait_tensorcnt(0);
            __syncthreads();
            lds_publish(&bstage[0][0]);
        }
    }

    const int cn  = lane & 15;
    const int cmh = (lane >> 4) << 3;
#pragma unroll
    for (int t = 0; t < 4; ++t) {
#pragma unroll
        for (int v = 0; v < 8; ++v) {
            const int m = m0 + v + cmh;
            const int n = n0 + t * 16 + cn;
            out[(size_t)m * DIM + n] = acc[t][v] + bo[n];
        }
    }
}

// ---------------------------------------------------------------------------
extern "C" void kernel_launch(void* const* d_in, const int* in_sizes, int n_in,
                              void* d_out, int out_size, void* d_ws,
                              size_t ws_size, hipStream_t stream) {
    const float* x  = (const float*)d_in[0];
    // d_in[1] = mask (recomputed analytically; unused)
    const float* Wq = (const float*)d_in[2];
    const float* bq = (const float*)d_in[3];
    const float* Wk = (const float*)d_in[4];
    const float* bk = (const float*)d_in[5];
    const float* Wv = (const float*)d_in[6];
    const float* bv = (const float*)d_in[7];
    const float* Wo = (const float*)d_in[8];
    const float* bo = (const float*)d_in[9];
    float* out = (float*)d_out;

    char* ws = (char*)d_ws;
    size_t off = 0;
    const size_t szX = (size_t)MTOT * DIM * sizeof(unsigned short); // 16 MB
    const size_t szW = (size_t)DIM * DIM * sizeof(unsigned short);  //  2 MB
    unsigned short* xb  = (unsigned short*)(ws + off); off += szX;
    unsigned short* wqb = (unsigned short*)(ws + off); off += szW;
    unsigned short* wkb = (unsigned short*)(ws + off); off += szW;
    unsigned short* wvb = (unsigned short*)(ws + off); off += szW;
    unsigned short* wob = (unsigned short*)(ws + off); off += szW;
    unsigned short* qbuf = (unsigned short*)(ws + off); off += szX;
    unsigned short* kbuf = (unsigned short*)(ws + off); off += szX;
    unsigned short* vtb  = (unsigned short*)(ws + off); off += szX;
    unsigned short* attb = (unsigned short*)(ws + off); off += szX;

    const int nX = MTOT * DIM;
    const int nW = DIM * DIM;
    convert_bf16<<<(nX + 255) / 256, 256, 0, stream>>>(x,  xb,  nX);
    convert_bf16<<<(nW + 255) / 256, 256, 0, stream>>>(Wq, wqb, nW);
    convert_bf16<<<(nW + 255) / 256, 256, 0, stream>>>(Wk, wkb, nW);
    convert_bf16<<<(nW + 255) / 256, 256, 0, stream>>>(Wv, wvb, nW);
    convert_bf16<<<(nW + 255) / 256, 256, 0, stream>>>(Wo, wob, nW);

    dim3 g1(MTOT / 64, HEADS, 3);
    qkv_gemm<<<g1, 128, 0, stream>>>(xb, wqb, wkb, wvb, bq, bk, bv,
                                     qbuf, kbuf, vtb);

    dim3 g2(SEQ / 64, BATCH * HEADS);
    flash_attn<<<g2, 128, 0, stream>>>(qbuf, kbuf, vtb, attb);

    dim3 g3(MTOT / 64, DIM / 64);
    out_gemm<<<g3, 128, 0, stream>>>(attb, wob, bo, out);
}
